// AdaHAN_66546223284715
// MI455X (gfx1250) — matrix operand order repaired and checked by reading, not compiled, not run on hardware
//
#include <hip/hip_runtime.h>
#include <math.h>

typedef __attribute__((ext_vector_type(16))) _Float16 v16h;
typedef __attribute__((ext_vector_type(8)))  _Float16 v8h;
typedef __attribute__((ext_vector_type(8)))  float    v8f;

#define SENT_LEN 2048

// ---------------------------------------------------------------------------
// Workspace layout.
// Float region (at offset 0, in floats):
//   ctmp : 8*256*256 = 524288   (NCHW f32 conv scratch, reused per layer)
//   x3   : 8*32*32   = 8192     (NCHW f32, input to fuse)
//   h    : 8
// Half region (starts at float offset 532488 -> byte 2129952, 32B aligned):
//   imgp : 262*260*8 = 544960   (padded NHWC f16 image, 8 ch, 3 real)
//   x1p  : 134*132*8 = 141504
//   x2p  : 70*68*8   = 38080
//   wpk0/1/2 : 16*224 = 3584 each (packed f16 weights, zero-padded)
// Padded buffers have H+6 rows: halo of 2 plus 2 extra rows because the
// K-padding taps (25..27, multiplied by zero weights) read up to row y+5.
// ---------------------------------------------------------------------------
#define CTMP_OFF   0
#define X3_OFF     (8 * 256 * 256)
#define H_OFF      (X3_OFF + 8 * 32 * 32)
#define HALF_BYTE_OFF ((size_t)(H_OFF + 8) * 4)   // 2129952
#define IMGP_OFF   0
#define X1P_OFF    (IMGP_OFF + 262 * 260 * 8)
#define X2P_OFF    (X1P_OFF + 134 * 132 * 8)
#define WPK0_OFF   (X2P_OFF + 70 * 68 * 8)
#define WPK1_OFF   (WPK0_OFF + 16 * 224)
#define WPK2_OFF   (WPK1_OFF + 16 * 224)

// ---------------------------------------------------------------------------
// 5x5 conv (pad 2) as implicit GEMM on v_wmma_f32_16x16x32_f16.
// K = tap*8 + ic, zero-padded to 224 (7 k-steps). Input is halo-padded NHWC
// f16, so every 8-consecutive-K group of the A fragment is one b128 load and
// there are NO bounds checks in the loop. B comes from pre-packed [16][224]
// f16 weights (rows 8..15 and k>=200 are zero).
// A layout: lane l (hi=l>>4, m=l&15) holds row M=m, elements j<8 -> K =
//   32ks + 8hi + j (tap 4ks+hi), j>=8 -> K = 32ks + 16 + 8hi + (j-8)
//   (tap 4ks+2+hi). B: lane column N=m, same K order as A.
// D: lane holds channel m, pixels x0 + 8*hi + 0..7 (contiguous f32 stores).
// ---------------------------------------------------------------------------
template <int W>
__device__ __forceinline__ constexpr int tapoff(int t) {
  return ((t / 5) * (W + 4) + (t % 5)) * 8;   // offset in halves
}

template <int H, int W>
__device__ __forceinline__ void conv5x5_wmma_seg(
    const _Float16* __restrict__ inp,   // padded NHWC f16 [(H+6)][(W+4)][8]
    const _Float16* __restrict__ wpk,   // packed [16][224] f16
    const float* __restrict__ bias,
    float* __restrict__ out,            // NCHW f32 [8][H][W]
    int seg, int lane)
{
  const int segsPerRow = W >> 4;
  const int y  = seg / segsPerRow;
  const int x0 = (seg - y * segsPerRow) << 4;
  const int m  = lane & 15;
  const int hi = lane >> 4;

  const _Float16* abase = inp + ((size_t)y * (W + 4) + x0 + m) * 8;
  const _Float16* bbase = wpk + (size_t)m * 224 + hi * 8;

  v8f acc = {0.f, 0.f, 0.f, 0.f, 0.f, 0.f, 0.f, 0.f};
#pragma unroll
  for (int ks = 0; ks < 7; ++ks) {
    const int o0 = hi ? tapoff<W>(4 * ks + 1) : tapoff<W>(4 * ks + 0);
    const int o1 = hi ? tapoff<W>(4 * ks + 3) : tapoff<W>(4 * ks + 2);
    v8h a0 = *(const v8h*)(abase + o0);
    v8h a1 = *(const v8h*)(abase + o1);
    v8h b0 = *(const v8h*)(bbase + ks * 32);
    v8h b1 = *(const v8h*)(bbase + ks * 32 + 16);
    v16h a = __builtin_shufflevector(a0, a1, 0,1,2,3,4,5,6,7,8,9,10,11,12,13,14,15);
    v16h b = __builtin_shufflevector(b0, b1, 0,1,2,3,4,5,6,7,8,9,10,11,12,13,14,15);
    acc = __builtin_amdgcn_wmma_f32_16x16x32_f16(false, a, false, b,
                                                 (short)0, acc, false, false);
  }

  if (m < 8) {
    const float bb = bias[m];
    float* dst = out + ((size_t)m * H + y) * W + x0 + 8 * hi;
#pragma unroll
    for (int r = 0; r < 8; ++r) dst[r] = acc[r] + bb;
  }
}

template <int H, int W>
__global__ void __launch_bounds__(256)
conv5x5_kernel(const _Float16* __restrict__ inp, const _Float16* __restrict__ wpk,
               const float* __restrict__ bias, float* __restrict__ out)
{
  const int wave = threadIdx.x >> 5;
  const int lane = threadIdx.x & 31;
  conv5x5_wmma_seg<H, W>(inp, wpk, bias, out, blockIdx.x * 8 + wave, lane);
}

// ---------------------------------------------------------------------------
// Prep: image NCHW f32 -> padded NHWC f16 (ch 0..2 real, 3..7 zero).
// ---------------------------------------------------------------------------
__global__ void __launch_bounds__(256)
pack_image(const float* __restrict__ img, _Float16* __restrict__ out)
{
  const int idx = blockIdx.x * 256 + threadIdx.x;   // padded pixel id (260x260)
  if (idx >= 260 * 260) return;
  const int py = idx / 260, px = idx - py * 260;
  const int oy = py - 2,    ox = px - 2;
  v8h v = {(_Float16)0, (_Float16)0, (_Float16)0, (_Float16)0,
           (_Float16)0, (_Float16)0, (_Float16)0, (_Float16)0};
  if (oy >= 0 && oy < 256 && ox >= 0 && ox < 256) {
    const int p = oy * 256 + ox;
    v[0] = (_Float16)img[p];
    v[1] = (_Float16)img[65536 + p];
    v[2] = (_Float16)img[131072 + p];
  }
  *(v8h*)(out + (size_t)idx * 8) = v;
}

// Weights [8][CIN][5][5] f32 -> [16][224] f16 with K = tap*8 + ic.
__global__ void __launch_bounds__(256)
pack_weights(const float* __restrict__ w, int cin, _Float16* __restrict__ dst)
{
  const int idx = blockIdx.x * 256 + threadIdx.x;   // 16*224 = 3584
  if (idx >= 16 * 224) return;
  const int n = idx / 224, k = idx - n * 224;
  const int tap = k >> 3, ic = k & 7;
  float v = 0.f;
  if (n < 8 && tap < 25 && ic < cin) v = w[((size_t)n * cin + ic) * 25 + tap];
  dst[idx] = (_Float16)v;
}

// ---------------------------------------------------------------------------
// Stage 1: conv1 blocks run concurrently with one GRU block; the GRU's 2048
// sequential steps are the latency pole, conv1 rides under it for free.
// ---------------------------------------------------------------------------
__global__ void __launch_bounds__(256)
stage1_conv1_gru(const _Float16* __restrict__ imgp, const _Float16* __restrict__ wpk0,
                 const float* __restrict__ c1b, float* __restrict__ conv_out,
                 const float* __restrict__ emb, const int* __restrict__ sent,
                 const float* __restrict__ w_ih, const float* __restrict__ w_hh,
                 const float* __restrict__ b_ih, const float* __restrict__ b_hh,
                 float* __restrict__ h_out, int nConvBlocks)
{
  __shared__ __align__(16) float xs[SENT_LEN * 8];  // 64 KB of 320 KB LDS

  if ((int)blockIdx.x < nConvBlocks) {   // uniform branch: EXEC stays all-1s
    const int wave = threadIdx.x >> 5;
    const int lane = threadIdx.x & 31;
    conv5x5_wmma_seg<256, 256>(imgp, wpk0, c1b, conv_out,
                               blockIdx.x * 8 + wave, lane);
    return;
  }

  // ---- GRU role: gather the whole embedded sentence into LDS, then one
  // wave runs the recurrence with lanes 0..23 owning gate rows. ----
  const int tid = threadIdx.x;
  for (int i = tid; i < SENT_LEN; i += 256) {
    const int row = sent[i];
    const float4* src = (const float4*)(emb + (size_t)row * 8);
    float4 lo = src[0], hi4 = src[1];
    float4* dst = (float4*)(&xs[i * 8]);
    dst[0] = lo; dst[1] = hi4;
  }
  __syncthreads();
  if (tid >= 32) return;

  const int l = tid;
  float wih[8], whh[8], bi = 0.f, bh = 0.f;
#pragma unroll
  for (int c = 0; c < 8; ++c) { wih[c] = 0.f; whh[c] = 0.f; }
  if (l < 24) {
#pragma unroll
    for (int c = 0; c < 8; ++c) { wih[c] = w_ih[l * 8 + c]; whh[c] = w_hh[l * 8 + c]; }
    bi = b_ih[l]; bh = b_hh[l];
  }
  float h[8];
#pragma unroll
  for (int c = 0; c < 8; ++c) h[c] = 0.f;

  for (int t = 0; t < SENT_LEN; ++t) {
    float gi = bi, gh = bh;
#pragma unroll
    for (int c = 0; c < 8; ++c) {
      gi = fmaf(wih[c], xs[t * 8 + c], gi);
      gh = fmaf(whh[c], h[c], gh);
    }
    const float s    = gi + gh;
    const float s8   = __shfl(s,  (l + 8)  & 31, 32);
    const float gi16 = __shfl(gi, (l + 16) & 31, 32);
    const float gh16 = __shfl(gh, (l + 16) & 31, 32);
    const float r = 1.f / (1.f + __expf(-s));
    const float z = 1.f / (1.f + __expf(-s8));
    const float n = tanhf(gi16 + r * gh16);
    const float hn = (1.f - z) * n + z * h[l & 7];
#pragma unroll
    for (int c = 0; c < 8; ++c) h[c] = __shfl(hn, c, 32);
  }
  if (l < 8) h_out[l] = h[l];
}

// ---------------------------------------------------------------------------
// AvgPool(3,2,1,count_include_pad)+ReLU, NCHW f32 in -> padded NHWC f16 out.
// Halo stays zero (buffer is memset before).
// ---------------------------------------------------------------------------
__global__ void __launch_bounds__(256)
pool3s2_relu_nhwc(const float* __restrict__ in, _Float16* __restrict__ out,
                  int H, int W)
{
  const int Hp = H >> 1, Wp = W >> 1;
  const int idx = blockIdx.x * 256 + threadIdx.x;
  if (idx >= 8 * Hp * Wp) return;
  const int c   = idx / (Hp * Wp);
  const int rem = idx - c * (Hp * Wp);
  const int py  = rem / Wp, px = rem - py * Wp;
  float s = 0.f;
#pragma unroll
  for (int dy = -1; dy <= 1; ++dy)
#pragma unroll
    for (int dx = -1; dx <= 1; ++dx) {
      const int iy = 2 * py + dy, ix = 2 * px + dx;
      if (iy >= 0 && iy < H && ix >= 0 && ix < W)
        s += in[((size_t)c * H + iy) * W + ix];
    }
  const float v = s * (1.f / 9.f);
  out[((size_t)(py + 2) * (Wp + 4) + (px + 2)) * 8 + c] =
      (_Float16)(v > 0.f ? v : 0.f);
}

// Last pool keeps NCHW f32 for the fusion kernel.
__global__ void __launch_bounds__(256)
pool3s2_relu_f32(const float* __restrict__ in, float* __restrict__ out,
                 int H, int W)
{
  const int Hp = H >> 1, Wp = W >> 1;
  const int idx = blockIdx.x * 256 + threadIdx.x;
  if (idx >= 8 * Hp * Wp) return;
  const int c   = idx / (Hp * Wp);
  const int rem = idx - c * (Hp * Wp);
  const int py  = rem / Wp, px = rem - py * Wp;
  float s = 0.f;
#pragma unroll
  for (int dy = -1; dy <= 1; ++dy)
#pragma unroll
    for (int dx = -1; dx <= 1; ++dx) {
      const int iy = 2 * py + dy, ix = 2 * px + dx;
      if (iy >= 0 && iy < H && ix >= 0 && ix < W)
        s += in[((size_t)c * H + iy) * W + ix];
    }
  const float v = s * (1.f / 9.f);
  out[idx] = v > 0.f ? v : 0.f;
}

// ---------------------------------------------------------------------------
// Fusion: h broadcast-add, 1x1 conv, L2 presence, top-2 (tie -> lower index),
// mask, 2-sparse FC (read only 2 of 1024 columns of fc1_w), log-softmax.
// ---------------------------------------------------------------------------
__global__ void __launch_bounds__(1024)
fuse_kernel(const float* __restrict__ x3, const float* __restrict__ hvec,
            const float* __restrict__ w11, const float* __restrict__ b11,
            const float* __restrict__ fcw, const float* __restrict__ fcb,
            float* __restrict__ out)
{
  __shared__ float s_m[1024];
  __shared__ float s_v1[1024], s_v2[1024];
  __shared__ int   s_i1[1024], s_i2[1024];
  __shared__ float s_red[1024];

  const int t = threadIdx.x;
  float e[8];
#pragma unroll
  for (int c = 0; c < 8; ++c) e[c] = x3[c * 1024 + t] + hvec[c];
  float r0 = b11[0], r1 = b11[1];
#pragma unroll
  for (int c = 0; c < 8; ++c) {
    r0 = fmaf(w11[c],     e[c], r0);
    r1 = fmaf(w11[8 + c], e[c], r1);
  }
  const float pres = r0 * r0 + r1 * r1;
  s_m[t]  = r0 + r1;
  s_v1[t] = pres;     s_i1[t] = t;
  s_v2[t] = -3.4e38f; s_i2[t] = 0x7fffffff;
  __syncthreads();

  for (int off = 512; off > 0; off >>= 1) {
    if (t < off) {
      float v1 = s_v1[t], v2 = s_v2[t];
      int   i1 = s_i1[t], i2 = s_i2[t];
#pragma unroll
      for (int q = 0; q < 2; ++q) {
        float cv = q ? s_v2[t + off] : s_v1[t + off];
        int   ci = q ? s_i2[t + off] : s_i1[t + off];
        if (cv > v1 || (cv == v1 && ci < i1)) {
          float tv = v1; int ti = i1; v1 = cv; i1 = ci; cv = tv; ci = ti;
        }
        if (cv > v2 || (cv == v2 && ci < i2)) { v2 = cv; i2 = ci; }
      }
      s_v1[t] = v1; s_i1[t] = i1; s_v2[t] = v2; s_i2[t] = i2;
    }
    __syncthreads();
  }
  const int   i1 = s_i1[0], i2 = s_i2[0];
  const float a1 = s_m[i1], a2 = s_m[i2];
  out[1000 + t] = (t == i1 || t == i2) ? 1.f : 0.f;

  float logit = -3.4e38f;
  if (t < 1000)
    logit = fcb[t] + a1 * fcw[(size_t)t * 1024 + i1]
                   + a2 * fcw[(size_t)t * 1024 + i2];
  s_red[t] = logit;
  __syncthreads();
  for (int off = 512; off > 0; off >>= 1) {
    if (t < off) { float o = s_red[t + off]; if (o > s_red[t]) s_red[t] = o; }
    __syncthreads();
  }
  const float mx = s_red[0];
  __syncthreads();
  s_red[t] = (t < 1000) ? __expf(logit - mx) : 0.f;
  __syncthreads();
  for (int off = 512; off > 0; off >>= 1) {
    if (t < off) s_red[t] += s_red[t + off];
    __syncthreads();
  }
  const float lse = __logf(s_red[0]);
  if (t < 1000) out[t] = logit - mx - lse;
}

// ---------------------------------------------------------------------------
extern "C" void kernel_launch(void* const* d_in, const int* in_sizes, int n_in,
                              void* d_out, int out_size, void* d_ws, size_t ws_size,
                              hipStream_t stream)
{
  (void)in_sizes; (void)n_in; (void)out_size; (void)ws_size;
  const float* image = (const float*)d_in[0];
  const int*   sent  = (const int*)  d_in[1];
  const float* c1w = (const float*)d_in[2];  const float* c1b = (const float*)d_in[3];
  const float* c2w = (const float*)d_in[4];  const float* c2b = (const float*)d_in[5];
  const float* c3w = (const float*)d_in[6];  const float* c3b = (const float*)d_in[7];
  const float* emb = (const float*)d_in[8];
  const float* wih = (const float*)d_in[9];  const float* whh = (const float*)d_in[10];
  const float* bih = (const float*)d_in[11]; const float* bhh = (const float*)d_in[12];
  const float* w11 = (const float*)d_in[13]; const float* b11 = (const float*)d_in[14];
  const float* fcw = (const float*)d_in[15]; const float* fcb = (const float*)d_in[16];

  float*    wsf  = (float*)d_ws;
  float*    ctmp = wsf + CTMP_OFF;
  float*    x3   = wsf + X3_OFF;
  float*    hb   = wsf + H_OFF;
  _Float16* wsh  = (_Float16*)((char*)d_ws + HALF_BYTE_OFF);
  _Float16* imgp = wsh + IMGP_OFF;
  _Float16* x1p  = wsh + X1P_OFF;
  _Float16* x2p  = wsh + X2P_OFF;
  _Float16* wpk0 = wsh + WPK0_OFF;
  _Float16* wpk1 = wsh + WPK1_OFF;
  _Float16* wpk2 = wsh + WPK2_OFF;

  // Zero the padded NHWC buffers (halo + K-padding overrun rows).
  hipMemsetAsync(imgp, 0, (size_t)262 * 260 * 8 * 2, stream);
  hipMemsetAsync(x1p,  0, (size_t)134 * 132 * 8 * 2, stream);
  hipMemsetAsync(x2p,  0, (size_t)70  * 68  * 8 * 2, stream);

  // Prep: pack image + weights into WMMA-friendly f16 layouts.
  pack_image<<<(260 * 260 + 255) / 256, 256, 0, stream>>>(image, imgp);
  pack_weights<<<14, 256, 0, stream>>>(c1w, 3, wpk0);
  pack_weights<<<14, 256, 0, stream>>>(c2w, 8, wpk1);
  pack_weights<<<14, 256, 0, stream>>>(c3w, 8, wpk2);

  // conv1 (4096 wave-tiles over 512 blocks) overlapped with the GRU block.
  stage1_conv1_gru<<<513, 256, 0, stream>>>(imgp, wpk0, c1b, ctmp,
                                            emb, sent, wih, whh, bih, bhh,
                                            hb, 512);
  pool3s2_relu_nhwc<<<(8 * 128 * 128) / 256, 256, 0, stream>>>(ctmp, x1p, 256, 256);
  conv5x5_kernel<128, 128><<<128, 256, 0, stream>>>(x1p, wpk1, c2b, ctmp);
  pool3s2_relu_nhwc<<<(8 * 64 * 64) / 256, 256, 0, stream>>>(ctmp, x2p, 128, 128);
  conv5x5_kernel<64, 64><<<32, 256, 0, stream>>>(x2p, wpk2, c3b, ctmp);
  pool3s2_relu_f32<<<(8 * 32 * 32) / 256, 256, 0, stream>>>(ctmp, x3, 64, 64);
  fuse_kernel<<<1, 1024, 0, stream>>>(x3, hb, w11, b11, fcw, fcb, (float*)d_out);
}
